// CapsuleNetwork_18562848653538
// MI455X (gfx1250) — compile-verified, matrix operand order checked by reading, not run on hardware
//
#include <hip/hip_runtime.h>
#include <hip/hip_bf16.h>

typedef __attribute__((ext_vector_type(16))) _Float16 v16h;
typedef __attribute__((ext_vector_type(8)))  float    v8f;

#define DEV __device__ __forceinline__

// ---------------- dimensions ----------------
constexpr int BATCH = 4;
constexpr int C1 = 256, H1 = 52, W1 = 52;      // conv1 output [4,256,52,52]
constexpr int KGEMM = 256 * 81;                // 20736 (im2col K)
constexpr int MGEMM = 512;                     // caps-conv output channels
constexpr int NCOLS = BATCH * 11 * 11;         // 484 real GEMM columns
constexpr int NPAD  = 512;                     // padded to multiple of 32
constexpr int NCAP2 = 7744;                    // routing-1 N
constexpr int M2 = 48, O2 = 24;                // routing-1 out caps / dim
constexpr int N3 = 48, M3 = 40, O3 = 32;       // routing-2

// ---------------- workspace layout (bytes) ----------------
constexpr size_t alignup(size_t x) { return (x + 255) & ~size_t(255); }
// Phase A (dead after squash):
constexpr size_t H16_OFF   = 0;                                                      // f16 [4*256*52*52]
constexpr size_t CW16_OFF  = alignup(H16_OFF  + 2ull * BATCH * C1 * H1 * W1);        // f16 [512][20736]
constexpr size_t IM2_OFF   = alignup(CW16_OFF + 2ull * MGEMM * KGEMM);               // f16 [512][20736]
constexpr size_t XPRE_OFF  = alignup(IM2_OFF  + 2ull * NPAD * KGEMM);                // f32 [4][7744][8]
// Phase B overlays phase A (x_hat2 written after xpre consumed):
constexpr size_t XHAT2_OFF = 0;                                                      // f32 [48][4][7744][24]
constexpr size_t XCAPS_OFF = alignup(XHAT2_OFF + 4ull * M2 * 4 * NCAP2 * O2);        // f32 [4][7744][8]
constexpr size_t B2_OFF    = alignup(XCAPS_OFF + 4ull * BATCH * NCAP2 * 8);          // f32 [48][4][7744]
constexpr size_t C2_OFF    = alignup(B2_OFF    + 4ull * M2 * 4 * NCAP2);
constexpr size_t V2_OFF    = alignup(C2_OFF    + 4ull * M2 * 4 * NCAP2);             // f32 [48][4][24]
constexpr size_t V2T_OFF   = alignup(V2_OFF    + 4ull * M2 * 4 * O2);                // f32 [4][48][24]
constexpr size_t XHAT3_OFF = alignup(V2T_OFF   + 4ull * M2 * 4 * O2);                // f32 [40][4][48][32]
constexpr size_t B3_OFF    = alignup(XHAT3_OFF + 4ull * M3 * 4 * N3 * O3);
constexpr size_t C3_OFF    = alignup(B3_OFF    + 4ull * M3 * 4 * N3);
constexpr size_t V3_OFF    = alignup(C3_OFF    + 4ull * M3 * 4 * N3);                // f32 [40][4][32]
constexpr size_t FLAT_OFF  = alignup(V3_OFF    + 4ull * M3 * 4 * O3);                // f32 [4][1280]
constexpr size_t FC1_OFF   = alignup(FLAT_OFF  + 4ull * 4 * 1280);                   // f32 [4][100]
constexpr size_t R1_OFF    = alignup(FC1_OFF   + 4ull * 4 * 100);                    // f32 [4][100]
constexpr size_t R2_OFF    = alignup(R1_OFF    + 4ull * 4 * 100);                    // f32 [4][1280]
constexpr size_t R3_OFF    = alignup(R2_OFF    + 4ull * 4 * 1280);                   // f32 [4][2048]
constexpr size_t R4_OFF    = alignup(R3_OFF    + 4ull * 4 * 2048);                   // f32 [4][4096]

// ---------------- conv1 + relu -> f16 ----------------
__global__ void conv1_relu_f16(const float* __restrict__ x, const float* __restrict__ w,
                               const float* __restrict__ bias, _Float16* __restrict__ h) {
  int idx = blockIdx.x * 256 + threadIdx.x;
  constexpr int TOT = BATCH * C1 * H1 * W1;
  if (idx >= TOT) return;
  int ox = idx % W1; int t = idx / W1;
  int oy = t % H1;  t /= H1;
  int oc = t % C1;  int b = t / C1;
  const float* xb = x + (size_t)b * 112 * 112;
  const float* wr = w + (size_t)oc * 81;
  float acc = bias[oc];
  int iy0 = oy * 2, ix0 = ox * 2;
#pragma unroll
  for (int ky = 0; ky < 9; ky++) {
    const float* xr = xb + (iy0 + ky) * 112 + ix0;
#pragma unroll
    for (int kx = 0; kx < 9; kx++) acc = fmaf(xr[kx], wr[ky * 9 + kx], acc);
  }
  h[idx] = (_Float16)fmaxf(acc, 0.f);
}

// ---------------- f32 -> f16 copy (caps weights; layout already K-major) ----------------
__global__ void f32_to_f16(const float* __restrict__ src, _Float16* __restrict__ dst, int n) {
  int idx = blockIdx.x * 256 + threadIdx.x;
  if (idx < n) dst[idx] = (_Float16)src[idx];
}

// ---------------- im2col for caps conv (stride 4, k9), K-major rows ----------------
__global__ void im2col_caps(const _Float16* __restrict__ h, _Float16* __restrict__ im2) {
  int idx = blockIdx.x * 256 + threadIdx.x;
  constexpr int TOT = NPAD * KGEMM;
  if (idx >= TOT) return;
  int k = idx % KGEMM; int col = idx / KGEMM;   // col = b*121 + y*11 + x
  _Float16 v = (_Float16)0.f;
  if (col < NCOLS) {
    int xx = col % 11; int t = col / 11; int yy = t % 11; int b = t / 11;
    int kx = k % 9; int t2 = k / 9; int ky = t2 % 9; int c = t2 / 9;
    v = h[(((size_t)b * C1 + c) * H1 + (yy * 4 + ky)) * W1 + (xx * 4 + kx)];
  }
  im2[idx] = v;
}

// ---------------- WMMA GEMM: C[512][512] = W[512][K] * im2col[N][K]^T ----------------
DEV v16h load_frag16(const _Float16* __restrict__ base, int row0, int k0) {
  // 16-bit A/B fragment, K-major source. Lanes 0-15: row=lane, K {0..7,16..23};
  // lanes 16-31: row=lane-16, K {8..15,24..31}.  (ISA 7.12.2)
  int lane = threadIdx.x & 31;
  const _Float16* p = base + (size_t)(row0 + (lane & 15)) * KGEMM + k0 + ((lane >> 4) << 3);
  v16h f;
  reinterpret_cast<float4*>(&f)[0] = *reinterpret_cast<const float4*>(p);
  reinterpret_cast<float4*>(&f)[1] = *reinterpret_cast<const float4*>(p + 16);
  return f;
}

DEV v8f wmma16(v16h a, v16h b, v8f c) {
  return __builtin_amdgcn_wmma_f32_16x16x32_f16(false, a, false, b, (short)0, c, false, false);
}

DEV void store_tile(v8f acc, int m0, int n0, const float* __restrict__ bias,
                    float* __restrict__ xpre) {
  int lane = threadIdx.x & 31;
  int n = n0 + (lane & 15);
  if (n >= NCOLS) return;
  int xx = n % 11; int t = n / 11; int yy = t % 11; int b = t / 11;
  int mbase = m0 + ((lane >> 4) << 3);
#pragma unroll
  for (int j = 0; j < 8; j++) {
    int m = mbase + j;                     // channel = cap*8 + r
    int cap = m >> 3, r = m & 7;
    size_t dst = ((size_t)b * 7744 + (size_t)yy * 704 + cap * 11 + xx) * 8 + r;
    xpre[dst] = ((const float*)&acc)[j] + bias[m];
  }
}

__global__ void caps_gemm_wmma(const _Float16* __restrict__ Wm, const _Float16* __restrict__ Bm,
                               const float* __restrict__ bias, float* __restrict__ xpre) {
  int m0 = blockIdx.y * 32;
  int n0 = blockIdx.x * 32;
  int lane = threadIdx.x & 31;
  v8f a00{}, a01{}, a10{}, a11{};
  for (int k = 0; k < KGEMM; k += 32) {
    __builtin_prefetch(Wm + (size_t)(m0 + (lane & 15)) * KGEMM + k + 512, 0, 1);
    __builtin_prefetch(Bm + (size_t)(n0 + (lane & 15)) * KGEMM + k + 512, 0, 1);
    v16h af0 = load_frag16(Wm, m0,      k);
    v16h af1 = load_frag16(Wm, m0 + 16, k);
    v16h bf0 = load_frag16(Bm, n0,      k);
    v16h bf1 = load_frag16(Bm, n0 + 16, k);
    a00 = wmma16(af0, bf0, a00);
    a01 = wmma16(af0, bf1, a01);
    a10 = wmma16(af1, bf0, a10);
    a11 = wmma16(af1, bf1, a11);
  }
  store_tile(a00, m0,      n0,      bias, xpre);
  store_tile(a01, m0,      n0 + 16, bias, xpre);
  store_tile(a10, m0 + 16, n0,      bias, xpre);
  store_tile(a11, m0 + 16, n0 + 16, bias, xpre);
}

// ---------------- squash over trailing dim 8 ----------------
__global__ void squash_dim8(const float* __restrict__ xpre, float* __restrict__ xc) {
  int idx = blockIdx.x * 256 + threadIdx.x;
  if (idx >= BATCH * NCAP2) return;
  const float* p = xpre + (size_t)idx * 8;
  float v[8]; float n2 = 0.f;
#pragma unroll
  for (int i = 0; i < 8; i++) { v[i] = p[i]; n2 += v[i] * v[i]; }
  float s = (n2 / (1.f + n2)) * rsqrtf(n2 + 1e-9f);
  float* o = xc + (size_t)idx * 8;
#pragma unroll
  for (int i = 0; i < 8; i++) o[i] = v[i] * s;
}

// ---------------- x_hat = einsum('bni,mnio->mbno') ----------------
template <int NI, int NO>
__global__ void einsum_xhat(const float* __restrict__ xin, const float* __restrict__ Wt,
                            float* __restrict__ xhat, int M, int N) {
  int idx = blockIdx.x * 256 + threadIdx.x;
  if (idx >= M * N) return;
  int m = idx / N, n = idx % N;
  const float* wp = Wt + ((size_t)m * N + n) * (NI * NO);
  float xv[4][NI];
#pragma unroll
  for (int b = 0; b < 4; b++)
#pragma unroll
    for (int i = 0; i < NI; i++) xv[b][i] = xin[((size_t)b * N + n) * NI + i];
  float acc[4][NO];
#pragma unroll
  for (int b = 0; b < 4; b++)
#pragma unroll
    for (int o = 0; o < NO; o++) acc[b][o] = 0.f;
  for (int i = 0; i < NI; i++) {
#pragma unroll
    for (int o = 0; o < NO; o++) {
      float wv = wp[i * NO + o];
#pragma unroll
      for (int b = 0; b < 4; b++) acc[b][o] = fmaf(xv[b][i], wv, acc[b][o]);
    }
  }
  for (int b = 0; b < 4; b++) {
    float* op = xhat + (((size_t)m * 4 + b) * N + n) * NO;
#pragma unroll
    for (int o = 0; o < NO; o++) op[o] = acc[b][o];
  }
}

// ---------------- softmax over N for each (m,b); logits constant over O ----------------
__global__ void softmax_n(const float* __restrict__ bbuf, float* __restrict__ cbuf, int N) {
  int mb = blockIdx.x;
  const float* in = bbuf + (size_t)mb * N;
  float* out = cbuf + (size_t)mb * N;
  __shared__ float red[256];
  int t = threadIdx.x;
  float mx = -3.4e38f;
  for (int n = t; n < N; n += 256) mx = fmaxf(mx, in[n]);
  red[t] = mx; __syncthreads();
  for (int s = 128; s > 0; s >>= 1) { if (t < s) red[t] = fmaxf(red[t], red[t + s]); __syncthreads(); }
  mx = red[0]; __syncthreads();
  float sum = 0.f;
  for (int n = t; n < N; n += 256) sum += __expf(in[n] - mx);
  red[t] = sum; __syncthreads();
  for (int s = 128; s > 0; s >>= 1) { if (t < s) red[t] += red[t + s]; __syncthreads(); }
  float inv = 1.f / red[0];
  for (int n = t; n < N; n += 256) out[n] = __expf(in[n] - mx) * inv;
}

// ---------------- v = squash(sum_n x_hat*c) per (m,b) ----------------
template <int NO>
__global__ void route_v(const float* __restrict__ xhat, const float* __restrict__ cbuf,
                        float* __restrict__ vout, int N) {
  int mb = blockIdx.x;
  const float* xh = xhat + (size_t)mb * N * NO;
  const float* c  = cbuf + (size_t)mb * N;
  __shared__ float s[NO];
  __shared__ float scl;
  int t = threadIdx.x;
  float acc[NO];
#pragma unroll
  for (int o = 0; o < NO; o++) acc[o] = 0.f;
  for (int n = t; n < N; n += 256) {
    float cv = c[n];
    const float* row = xh + (size_t)n * NO;
#pragma unroll
    for (int o = 0; o < NO; o++) acc[o] = fmaf(row[o], cv, acc[o]);
  }
  if (t < NO) s[t] = 0.f;
  __syncthreads();
#pragma unroll
  for (int o = 0; o < NO; o++) atomicAdd(&s[o], acc[o]);
  __syncthreads();
  if (t == 0) {
    float n2 = 0.f;
    for (int o = 0; o < NO; o++) n2 += s[o] * s[o];
    scl = (n2 / (1.f + n2)) * rsqrtf(n2 + 1e-9f);
  }
  __syncthreads();
  if (t < NO) vout[(size_t)mb * NO + t] = s[t] * scl;
}

// ---------------- b += sum_o x_hat*v ----------------
template <int NO>
__global__ void route_update(const float* __restrict__ xhat, const float* __restrict__ v,
                             float* __restrict__ bbuf, int total, int N) {
  int idx = blockIdx.x * 256 + threadIdx.x;
  if (idx >= total) return;
  int mb = idx / N;
  const float* row = xhat + (size_t)idx * NO;
  const float* vv  = v + (size_t)mb * NO;
  float d = 0.f;
#pragma unroll
  for (int o = 0; o < NO; o++) d = fmaf(row[o], vv[o], d);
  bbuf[idx] += d;
}

// ---------------- v:[M][4][O] -> vt:[4][M][O] ----------------
__global__ void transpose_v(const float* __restrict__ v, float* __restrict__ vt, int M, int O) {
  int idx = blockIdx.x * 256 + threadIdx.x;
  if (idx >= M * 4 * O) return;
  int o = idx % O; int t = idx / O; int b = t % 4; int m = t / 4;
  vt[((size_t)b * M + m) * O + o] = v[((size_t)m * 4 + b) * O + o];
}

__global__ void fill_zero(float* __restrict__ p, int n) {
  int i = blockIdx.x * 256 + threadIdx.x;
  if (i < n) p[i] = 0.f;
}

// ---------------- y[b][j] = act(x[b]·w[j] + bias[j]); one wave per j ----------------
template <int ACT>  // 0=none 1=relu 2=sigmoid
__global__ void linear4(const float* __restrict__ x, const float* __restrict__ w,
                        const float* __restrict__ bias, float* __restrict__ y, int K, int J) {
  int wid = (blockIdx.x * blockDim.x + threadIdx.x) >> 5;
  int lane = threadIdx.x & 31;
  if (wid >= J) return;
  const float* wr = w + (size_t)wid * K;
  float a0 = 0.f, a1 = 0.f, a2 = 0.f, a3 = 0.f;
  for (int k = lane; k < K; k += 32) {
    float wv = wr[k];
    a0 = fmaf(x[k], wv, a0);
    a1 = fmaf(x[K + k], wv, a1);
    a2 = fmaf(x[2 * K + k], wv, a2);
    a3 = fmaf(x[3 * K + k], wv, a3);
  }
#pragma unroll
  for (int off = 16; off > 0; off >>= 1) {
    a0 += __shfl_down(a0, off, 32);
    a1 += __shfl_down(a1, off, 32);
    a2 += __shfl_down(a2, off, 32);
    a3 += __shfl_down(a3, off, 32);
  }
  if (lane == 0) {
    float bj = bias[wid];
    float r0 = a0 + bj, r1 = a1 + bj, r2 = a2 + bj, r3 = a3 + bj;
    if (ACT == 1) { r0 = fmaxf(r0, 0.f); r1 = fmaxf(r1, 0.f); r2 = fmaxf(r2, 0.f); r3 = fmaxf(r3, 0.f); }
    if (ACT == 2) {
      r0 = 1.f / (1.f + __expf(-r0)); r1 = 1.f / (1.f + __expf(-r1));
      r2 = 1.f / (1.f + __expf(-r2)); r3 = 1.f / (1.f + __expf(-r3));
    }
    y[wid] = r0; y[J + wid] = r1; y[2 * J + wid] = r2; y[3 * J + wid] = r3;
  }
}

static inline int nblk(long n) { return (int)((n + 255) / 256); }

extern "C" void kernel_launch(void* const* d_in, const int* in_sizes, int n_in,
                              void* d_out, int out_size, void* d_ws, size_t ws_size,
                              hipStream_t stream) {
  const float* x      = (const float*)d_in[0];
  const float* conv_w = (const float*)d_in[1];
  const float* conv_b = (const float*)d_in[2];
  const float* caps_w = (const float*)d_in[3];
  const float* caps_b = (const float*)d_in[4];
  const float* W2     = (const float*)d_in[5];
  const float* W3     = (const float*)d_in[6];
  const float* fc1_w  = (const float*)d_in[7];
  const float* fc1_b  = (const float*)d_in[8];
  const float* fc2_w  = (const float*)d_in[9];
  const float* fc2_b  = (const float*)d_in[10];
  const float* d1_w   = (const float*)d_in[11];
  const float* d1_b   = (const float*)d_in[12];
  const float* d2_w   = (const float*)d_in[13];
  const float* d2_b   = (const float*)d_in[14];
  const float* d3_w   = (const float*)d_in[15];
  const float* d3_b   = (const float*)d_in[16];
  const float* d4_w   = (const float*)d_in[17];
  const float* d4_b   = (const float*)d_in[18];
  const float* d5_w   = (const float*)d_in[19];
  const float* d5_b   = (const float*)d_in[20];

  char* ws = (char*)d_ws;
  _Float16* h16   = (_Float16*)(ws + H16_OFF);
  _Float16* cw16  = (_Float16*)(ws + CW16_OFF);
  _Float16* im2   = (_Float16*)(ws + IM2_OFF);
  float* xpre  = (float*)(ws + XPRE_OFF);
  float* xhat2 = (float*)(ws + XHAT2_OFF);
  float* xcaps = (float*)(ws + XCAPS_OFF);
  float* b2    = (float*)(ws + B2_OFF);
  float* c2    = (float*)(ws + C2_OFF);
  float* v2    = (float*)(ws + V2_OFF);
  float* v2t   = (float*)(ws + V2T_OFF);
  float* xhat3 = (float*)(ws + XHAT3_OFF);
  float* b3    = (float*)(ws + B3_OFF);
  float* c3    = (float*)(ws + C3_OFF);
  float* v3    = (float*)(ws + V3_OFF);
  float* flat  = (float*)(ws + FLAT_OFF);
  float* fc1o  = (float*)(ws + FC1_OFF);
  float* r1    = (float*)(ws + R1_OFF);
  float* r2    = (float*)(ws + R2_OFF);
  float* r3    = (float*)(ws + R3_OFF);
  float* r4    = (float*)(ws + R4_OFF);
  float* probs = (float*)d_out;
  float* recon = (float*)d_out + 16;

  // --- conv1 + relu (f16 output) ---
  conv1_relu_f16<<<nblk((long)BATCH * C1 * H1 * W1), 256, 0, stream>>>(x, conv_w, conv_b, h16);
  // --- caps weights to f16 (already [oc][c*81+ky*9+kx] K-major) ---
  f32_to_f16<<<nblk((long)MGEMM * KGEMM), 256, 0, stream>>>(caps_w, cw16, MGEMM * KGEMM);
  // --- im2col (stride 4, k9), columns padded to 512 ---
  im2col_caps<<<nblk((long)NPAD * KGEMM), 256, 0, stream>>>(h16, im2);
  // --- caps conv as WMMA GEMM, epilogue fuses bias + capsule reshape ---
  {
    dim3 grid(NPAD / 32, MGEMM / 32);
    caps_gemm_wmma<<<grid, 32, 0, stream>>>(cw16, im2, caps_b, xpre);
  }
  // --- squash primary capsules ---
  squash_dim8<<<nblk((long)BATCH * NCAP2), 256, 0, stream>>>(xpre, xcaps);

  // --- routing level 2: x_hat then 3 routing iterations ---
  einsum_xhat<8, O2><<<nblk((long)M2 * NCAP2), 256, 0, stream>>>(xcaps, W2, xhat2, M2, NCAP2);
  fill_zero<<<nblk((long)M2 * 4 * NCAP2), 256, 0, stream>>>(b2, M2 * 4 * NCAP2);
  for (int it = 0; it < 3; ++it) {
    softmax_n<<<M2 * 4, 256, 0, stream>>>(b2, c2, NCAP2);
    route_v<O2><<<M2 * 4, 256, 0, stream>>>(xhat2, c2, v2, NCAP2);
    if (it < 2)
      route_update<O2><<<nblk((long)M2 * 4 * NCAP2), 256, 0, stream>>>(xhat2, v2, b2,
                                                                       M2 * 4 * NCAP2, NCAP2);
  }
  transpose_v<<<nblk((long)M2 * 4 * O2), 256, 0, stream>>>(v2, v2t, M2, O2);

  // --- routing level 3 ---
  einsum_xhat<O2, O3><<<nblk((long)M3 * N3), 256, 0, stream>>>(v2t, W3, xhat3, M3, N3);
  fill_zero<<<nblk((long)M3 * 4 * N3), 256, 0, stream>>>(b3, M3 * 4 * N3);
  for (int it = 0; it < 3; ++it) {
    softmax_n<<<M3 * 4, 256, 0, stream>>>(b3, c3, N3);
    route_v<O3><<<M3 * 4, 256, 0, stream>>>(xhat3, c3, v3, N3);
    if (it < 2)
      route_update<O3><<<nblk((long)M3 * 4 * N3), 256, 0, stream>>>(xhat3, v3, b3,
                                                                    M3 * 4 * N3, N3);
  }
  transpose_v<<<nblk((long)M3 * 4 * O3), 256, 0, stream>>>(v3, flat, M3, O3);

  // --- FC head + decoder (wave-per-row GEMV, 8 waves/block) ---
  linear4<1><<<(100 + 7) / 8, 256, 0, stream>>>(flat, fc1_w, fc1_b, fc1o, 1280, 100);
  linear4<0><<<1, 256, 0, stream>>>(fc1o, fc2_w, fc2_b, probs, 100, 4);
  linear4<1><<<(100 + 7) / 8, 256, 0, stream>>>(probs, d1_w, d1_b, r1, 4, 100);
  linear4<1><<<(1280 + 7) / 8, 256, 0, stream>>>(r1, d2_w, d2_b, r2, 100, 1280);
  linear4<1><<<(2048 + 7) / 8, 256, 0, stream>>>(r2, d3_w, d3_b, r3, 1280, 2048);
  linear4<1><<<(4096 + 7) / 8, 256, 0, stream>>>(r3, d4_w, d4_b, r4, 2048, 4096);
  linear4<2><<<(12544 + 7) / 8, 256, 0, stream>>>(r4, d5_w, d5_b, recon, 4096, 12544);
  (void)in_sizes; (void)n_in; (void)out_size; (void)ws_size;
}
